// SAConvLSTM_2714419331527
// MI455X (gfx1250) — matrix operand order repaired and checked by reading, not compiled, not run on hardware
//
#include <hip/hip_runtime.h>
#include <math.h>

// SA-ConvLSTM for MI455X (gfx1250, wave32, WMMA).
// All heavy GEMMs run on v_wmma_f32_16x16x32_f16 (f16 in, f32 accumulate).

typedef _Float16 half_t;
typedef __attribute__((ext_vector_type(16))) _Float16 v16h;
typedef __attribute__((ext_vector_type(8)))  float    v8f;

#define HW    1024
#define WIDTH 32
#define CIN   64
#define AHD   64
#define COUT  64
#define CZ    128     // concat(ax, h) channels
#define TT    16
#define BB    8

// ---------------- WMMA helpers (contiguous-k layouts: k = h + (lane>>4)*16 for A and B) ----------------

__device__ __forceinline__ v8f wmma16(v16h a, v16h b, v8f c) {
  // (neg_a, A, neg_b, B, c_mod, C, reuse_a, reuse_b)
  return __builtin_amdgcn_wmma_f32_16x16x32_f16(false, a, false, b, (short)0, c,
                                                false, false);
}

// A-matrix 16x32 f16: lane (m = lane&15), halves h -> K = h + (lane>>4)*16.
// p points at element (row 0, K 0) of a row-major matrix with leading dim `ld` halves.
__device__ __forceinline__ v16h load_a_frag(const half_t* p, int ld, int lane) {
  int m  = lane & 15;
  int c0 = (lane >> 4) << 4;
  const half_t* row = p + m * ld + c0;
  v16h a;
#pragma unroll
  for (int h = 0; h < 8; ++h) a[h] = row[h];
#pragma unroll
  for (int h = 0; h < 8; ++h) a[8 + h] = row[8 + h];
  return a;
}

// B-matrix 32x16 f16 built from source S[n][k] (row-major, ld halves, contiguous in k):
// lane: n = lane&15, halves h -> k = h + (lane>=16)*16.  => one contiguous 32B run per lane.
__device__ __forceinline__ v16h load_b_frag(const half_t* p, int ld, int lane) {
  int n  = lane & 15;
  int k0 = (lane >> 4) << 4;
  const half_t* row = p + n * ld + k0;
  v16h b;
#pragma unroll
  for (int h = 0; h < 16; ++h) b[h] = row[h];
  return b;
}

// ---------------- prep kernels ----------------

__global__ void k_init(half_t* Z, float* Cst, int nZ, int nC) {
  int i = blockIdx.x * blockDim.x + threadIdx.x;
  if (i < nZ) Z[i] = (half_t)0.f;
  if (i < nC) Cst[i] = 0.f;
}

// conv_w (256, 128, 3, 3) f32  ->  Wt (tap=dy*3+dx, oc, ci) f16
__global__ void k_pack_w(const float* conv_w, half_t* Wt) {
  int i = blockIdx.x * blockDim.x + threadIdx.x;
  if (i >= 9 * 256 * CZ) return;
  int ci  = i & 127;
  int oc  = (i >> 7) & 255;
  int tap = i >> 15;
  int dy = tap / 3, dx = tap % 3;
  Wt[i] = (half_t)conv_w[((oc * CZ + ci) * 3 + dy) * 3 + dx];
}

// ---------------- 1x1 QKV projection (small GEMM, VALU) ----------------
// Q,Kt stored (b, n, AH) f16 ; V stored (b, Cin, n) f16 — layouts feed WMMA frags directly.
__global__ void k_qkv(const float* __restrict__ X,
                      const float* __restrict__ qw, const float* __restrict__ qb,
                      const float* __restrict__ kw, const float* __restrict__ kb,
                      const float* __restrict__ vw, const float* __restrict__ vb,
                      half_t* __restrict__ Q, half_t* __restrict__ Kt,
                      half_t* __restrict__ V, int t) {
  __shared__ float xs[CIN][4];
  int b  = blockIdx.y;
  int nb = blockIdx.x * 4;
  int o  = threadIdx.x;   // 0..63 output channel
  int nn = threadIdx.y;   // 0..3 pixel within block
  int lt = threadIdx.y * 64 + threadIdx.x;
  {
    int c = lt >> 2, j = lt & 3;
    xs[c][j] = X[((size_t)(b * CIN + c) * TT + t) * HW + nb + j];
  }
  __syncthreads();
  float qa = qb[o], ka = kb[o], va = vb[o];
#pragma unroll 8
  for (int c = 0; c < CIN; ++c) {
    float xv = xs[c][nn];
    qa = fmaf(qw[o * CIN + c], xv, qa);
    ka = fmaf(kw[o * CIN + c], xv, ka);
    va = fmaf(vw[o * CIN + c], xv, va);
  }
  int n = nb + nn;
  Q [((size_t)b * HW + n) * AHD + o] = (half_t)qa;
  Kt[((size_t)b * HW + n) * AHD + o] = (half_t)ka;
  V [((size_t)b * CIN + o) * HW + n] = (half_t)va;
}

// ---------------- flash attention (WMMA) ----------------
// grid = B*8 blocks, block = 128 (4 waves). Wave handles 32 queries x all 1024 keys
// (two 16-row A tiles sharing every K/V B-fragment => 2x arithmetic intensity).
// Row sums of P come from an extra WMMA against an all-ones B matrix (no shuffle adds).
// Writes attention output into Z channels [0,64) in pixel-major (b, n, cz) f16.
__global__ __launch_bounds__(128) void k_attn(const half_t* __restrict__ Q,
                                              const half_t* __restrict__ Kt,
                                              const half_t* __restrict__ V,
                                              half_t* __restrict__ Z) {
  __shared__ half_t plds[4][2][16 * 32];  // per-wave P staging (C-layout -> A-layout)
  int wave = threadIdx.x >> 5;
  int lane = threadIdx.x & 31;
  int b     = blockIdx.x >> 3;
  int qt    = blockIdx.x & 7;
  int qBase = qt * 128 + wave * 32;
  const half_t* Qb = Q  + ((size_t)b * HW + qBase) * AHD;
  const half_t* Kb = Kt + (size_t)b * HW * AHD;
  const half_t* Vb = V  + (size_t)b * CIN * HW;

  // Q fragments: [qtile][AH half]
  v16h qa[2][2];
#pragma unroll
  for (int q = 0; q < 2; ++q) {
    qa[q][0] = load_a_frag(Qb + (size_t)q * 16 * AHD,      AHD, lane);
    qa[q][1] = load_a_frag(Qb + (size_t)q * 16 * AHD + 32, AHD, lane);
  }

  // all-ones B fragment for row-sum WMMA
  v16h ones;
#pragma unroll
  for (int h = 0; h < 16; ++h) ones[h] = (half_t)1.f;

  v8f oacc[2][4];
  float rmax[2][8], rsum[2][8];
#pragma unroll
  for (int q = 0; q < 2; ++q) {
#pragma unroll
    for (int i = 0; i < 4; ++i)
#pragma unroll
      for (int r = 0; r < 8; ++r) oacc[q][i][r] = 0.f;
#pragma unroll
    for (int r = 0; r < 8; ++r) { rmax[q][r] = -1e30f; rsum[q][r] = 0.f; }
  }

  int hi = lane >> 4;
  int ln = lane & 15;

  for (int mb = 0; mb < HW; mb += 32) {
    // K fragments for keys [mb,mb+16) and [mb+16,mb+32), AH halves 0/1 — shared by both q tiles
    v16h kb00 = load_b_frag(Kb + (size_t)mb * AHD,             AHD, lane);
    v16h kb01 = load_b_frag(Kb + (size_t)mb * AHD + 32,        AHD, lane);
    v16h kb10 = load_b_frag(Kb + (size_t)(mb + 16) * AHD,      AHD, lane);
    v16h kb11 = load_b_frag(Kb + (size_t)(mb + 16) * AHD + 32, AHD, lane);

    float scv[2][8];
#pragma unroll
    for (int q = 0; q < 2; ++q) {
      v8f s0, s1;
#pragma unroll
      for (int r = 0; r < 8; ++r) { s0[r] = 0.f; s1[r] = 0.f; }
      s0 = wmma16(qa[q][0], kb00, s0);
      s0 = wmma16(qa[q][1], kb01, s0);
      s1 = wmma16(qa[q][0], kb10, s1);
      s1 = wmma16(qa[q][1], kb11, s1);

      half_t* P = plds[wave][q];
      // online softmax: row max via 16-lane butterfly, exp, rescale O
#pragma unroll
      for (int r = 0; r < 8; ++r) {
        float v0 = s0[r], v1 = s1[r];
        float mx = fmaxf(v0, v1);
        mx = fmaxf(mx, __shfl_xor(mx, 1));
        mx = fmaxf(mx, __shfl_xor(mx, 2));
        mx = fmaxf(mx, __shfl_xor(mx, 4));
        mx = fmaxf(mx, __shfl_xor(mx, 8));
        float mnew = fmaxf(rmax[q][r], mx);
        float p0 = __expf(v0 - mnew);
        float p1 = __expf(v1 - mnew);
        float sc = __expf(rmax[q][r] - mnew);
        rmax[q][r] = mnew;
        scv[q][r] = sc;
#pragma unroll
        for (int i2 = 0; i2 < 4; ++i2) oacc[q][i2][r] *= sc;
        int m = r + hi * 8;                      // query row of this slot
        P[m * 32 + ln]      = (half_t)p0;        // key col mb+ln
        P[m * 32 + 16 + ln] = (half_t)p1;        // key col mb+16+ln
      }
    }
    asm volatile("s_wait_dscnt 0" ::: "memory");  // LDS transpose fence (per-wave)

    // P A-fragments + row sums via ones-WMMA, then O += P * V
    v16h pa[2];
#pragma unroll
    for (int q = 0; q < 2; ++q) {
      pa[q] = load_a_frag(plds[wave][q], 32, lane);
      v8f lblk;
#pragma unroll
      for (int r = 0; r < 8; ++r) lblk[r] = 0.f;
      lblk = wmma16(pa[q], ones, lblk);          // every column = row sum of P
#pragma unroll
      for (int r = 0; r < 8; ++r)
        rsum[q][r] = rsum[q][r] * scv[q][r] + lblk[r];
    }
#pragma unroll
    for (int i2 = 0; i2 < 4; ++i2) {
      v16h vbf = load_b_frag(Vb + (size_t)i2 * 16 * HW + mb, HW, lane);
#pragma unroll
      for (int q = 0; q < 2; ++q)
        oacc[q][i2] = wmma16(pa[q], vbf, oacc[q][i2]);
    }
  }

  // normalize and scatter into z-buffer channels 0..63 (pixel-major)
#pragma unroll
  for (int q = 0; q < 2; ++q)
#pragma unroll
    for (int r = 0; r < 8; ++r) {
      float inv = 1.f / rsum[q][r];
      int pix = qBase + q * 16 + r + hi * 8;
#pragma unroll
      for (int i2 = 0; i2 < 4; ++i2)
        Z[((size_t)b * HW + pix) * CZ + i2 * 16 + ln] =
            (half_t)(oacc[q][i2][r] * inv);
    }
}

// ---------------- 3x3 gate conv as 9 shifted GEMMs (WMMA) ----------------
// grid = B*32 row tiles, block = 128 (4 waves). Wave w: oc [w*64, w*64+64) x 32 pixels
// (two 16-pixel B tiles share every weight A-fragment => half the weight traffic).
__global__ __launch_bounds__(128) void k_conv(const half_t* __restrict__ Z,
                                              const half_t* __restrict__ Wt,
                                              float* __restrict__ G) {
  int wave = threadIdx.x >> 5;
  int lane = threadIdx.x & 31;
  int b  = blockIdx.x >> 5;
  int pt = blockIdx.x & 31;        // image row y = pt ; pixels [pt*32, pt*32+32)
  int pb = pt * 32;
  const half_t* Zb = Z + (size_t)b * HW * CZ;
  int ln = lane & 15;
  int hi = lane >> 4;
  int ocb = wave * 64;

  v8f acc0[4], acc1[4];
#pragma unroll
  for (int i = 0; i < 4; ++i)
#pragma unroll
    for (int r = 0; r < 8; ++r) { acc0[i][r] = 0.f; acc1[i][r] = 0.f; }

  for (int tap = 0; tap < 9; ++tap) {
    int dyo = tap / 3 - 1, dxo = tap % 3 - 1;
    int sy = pt + dyo;
    bool rowok = (sy >= 0) && (sy < 32);
    int sx0 = ln + dxo;            // tile 0: x = ln
    int sx1 = 16 + ln + dxo;       // tile 1: x = 16 + ln
    bool v0ok = rowok && (sx0 >= 0) && (sx0 < 32);
    bool v1ok = rowok && (sx1 >= 0) && (sx1 < 32);
    int sp0 = v0ok ? (sy * 32 + sx0) : 0;
    int sp1 = v1ok ? (sy * 32 + sx1) : 0;
    const half_t* zr0 = Zb + (size_t)sp0 * CZ + hi * 16;   // k = hi*16 + h
    const half_t* zr1 = Zb + (size_t)sp1 * CZ + hi * 16;
    const half_t* wt  = Wt + (size_t)tap * 256 * CZ;
#pragma unroll
    for (int cbk = 0; cbk < CZ; cbk += 32) {
      v16h bf0, bf1;
#pragma unroll
      for (int h = 0; h < 16; ++h) { bf0[h] = zr0[cbk + h]; bf1[h] = zr1[cbk + h]; }
      if (!v0ok) {
#pragma unroll
        for (int h = 0; h < 16; ++h) bf0[h] = (half_t)0.f;
      }
      if (!v1ok) {
#pragma unroll
        for (int h = 0; h < 16; ++h) bf1[h] = (half_t)0.f;
      }
#pragma unroll
      for (int i2 = 0; i2 < 4; ++i2) {
        v16h af = load_a_frag(wt + (size_t)(ocb + i2 * 16) * CZ + cbk, CZ, lane);
        acc0[i2] = wmma16(af, bf0, acc0[i2]);
        acc1[i2] = wmma16(af, bf1, acc1[i2]);
      }
    }
  }

  float* Gb = G + (size_t)b * 256 * HW;
#pragma unroll
  for (int i2 = 0; i2 < 4; ++i2)
#pragma unroll
    for (int r = 0; r < 8; ++r) {
      int oc = ocb + i2 * 16 + r + hi * 8;
      Gb[(size_t)oc * HW + pb + ln]      = acc0[i2][r];
      Gb[(size_t)oc * HW + pb + 16 + ln] = acc1[i2][r];
    }
}

// ---------------- LSTM gates (elementwise) ----------------
__global__ void k_gates(const float* __restrict__ G, const float* __restrict__ conv_b,
                        const float* __restrict__ wci, const float* __restrict__ wcf,
                        const float* __restrict__ wco,
                        float* __restrict__ Cst, half_t* __restrict__ Z,
                        float* __restrict__ out, int t) {
  int i = blockIdx.x * blockDim.x + threadIdx.x;
  if (i >= BB * COUT * HW) return;
  int p  = i & (HW - 1);
  int cc = (i >> 10) & 63;
  int b  = i >> 16;
  const float* Gb = G + (size_t)b * 256 * HW;
  float ic = Gb[(size_t)cc * HW + p]         + conv_b[cc];
  float fc = Gb[(size_t)(64 + cc) * HW + p]  + conv_b[64 + cc];
  float gc = Gb[(size_t)(128 + cc) * HW + p] + conv_b[128 + cc];
  float oc = Gb[(size_t)(192 + cc) * HW + p] + conv_b[192 + cc];
  float cprev = Cst[i];
  float ig = 1.f / (1.f + __expf(-(ic + wci[cc * HW + p] * cprev)));
  float fg = 1.f / (1.f + __expf(-(fc + wcf[cc * HW + p] * cprev)));
  float nc = fg * cprev + ig * tanhf(gc);
  float og = 1.f / (1.f + __expf(-(oc + wco[cc * HW + p] * nc)));
  float nh = og * tanhf(nc);
  Cst[i] = nc;
  Z[((size_t)b * HW + p) * CZ + 64 + cc] = (half_t)nh;          // h -> z for t+1
  out[(((size_t)b * COUT + cc) * TT + t) * HW + p] = nh;        // (B, Cout, T, H, W)
}

// ---------------- host-side launch sequence ----------------

extern "C" void kernel_launch(void* const* d_in, const int* in_sizes, int n_in,
                              void* d_out, int out_size, void* d_ws, size_t ws_size,
                              hipStream_t stream) {
  const float* X   = (const float*)d_in[0];
  const float* qw  = (const float*)d_in[1];
  const float* qb  = (const float*)d_in[2];
  const float* kw  = (const float*)d_in[3];
  const float* kb  = (const float*)d_in[4];
  const float* vw  = (const float*)d_in[5];
  const float* vb  = (const float*)d_in[6];
  const float* cw  = (const float*)d_in[7];
  const float* cbi = (const float*)d_in[8];
  const float* wci = (const float*)d_in[9];
  const float* wcf = (const float*)d_in[10];
  const float* wco = (const float*)d_in[11];
  float* out = (float*)d_out;

  char* ws = (char*)d_ws;
  size_t off = 0;
  auto carve = [&](size_t bytes) -> void* {
    void* p = ws + off;
    off += (bytes + 255) & ~(size_t)255;
    return p;
  };
  half_t* Q   = (half_t*)carve((size_t)BB * HW * AHD * sizeof(half_t));   // 1 MB
  half_t* Kt  = (half_t*)carve((size_t)BB * HW * AHD * sizeof(half_t));   // 1 MB
  half_t* V   = (half_t*)carve((size_t)BB * CIN * HW * sizeof(half_t));   // 1 MB
  half_t* Z   = (half_t*)carve((size_t)BB * HW * CZ  * sizeof(half_t));   // 2 MB
  float*  Cst = (float*) carve((size_t)BB * COUT * HW * sizeof(float));   // 2 MB
  float*  G   = (float*) carve((size_t)BB * 256 * HW * sizeof(float));    // 8 MB
  half_t* Wt  = (half_t*)carve((size_t)9 * 256 * CZ * sizeof(half_t));    // ~0.6 MB

  const int nZ = BB * HW * CZ;        // 1,048,576 halves
  const int nC = BB * COUT * HW;      // 524,288 floats
  k_init<<<(nZ + 255) / 256, 256, 0, stream>>>(Z, Cst, nZ, nC);
  k_pack_w<<<(9 * 256 * CZ + 255) / 256, 256, 0, stream>>>(cw, Wt);

  for (int t = 0; t < TT; ++t) {
    k_qkv <<<dim3(HW / 4, BB), dim3(64, 4), 0, stream>>>(X, qw, qb, kw, kb, vw, vb,
                                                         Q, Kt, V, t);
    k_attn<<<BB * 8, 128, 0, stream>>>(Q, Kt, V, Z);
    k_conv<<<BB * 32, 128, 0, stream>>>(Z, Wt, G);
    k_gates<<<(BB * COUT * HW + 255) / 256, 256, 0, stream>>>(G, cbi, wci, wcf, wco,
                                                              Cst, Z, out, t);
  }
}
